// ComplexPolarAttention_36395552866681
// MI455X (gfx1250) — matrix operand order repaired and checked by reading, not compiled, not run on hardware
//
#include <hip/hip_runtime.h>

#define N_NODES 8192
#define D_HID   64
#define N_EDGES 262144
#define DQK     128   // [c | s]
#define DV      128   // [mag | phase]

typedef __attribute__((ext_vector_type(16))) _Float16 v16h;
typedef __attribute__((ext_vector_type(8)))  _Float16 v8h;
typedef __attribute__((ext_vector_type(8)))  float    v8f;

static __device__ __forceinline__ v16h cat8(v8h lo, v8h hi) {
  return __builtin_shufflevector(lo, hi, 0,1,2,3,4,5,6,7,8,9,10,11,12,13,14,15);
}

// ---------------------------------------------------------------------------
// Kernel 1: build Q/K = [mag*cos(phase) | mag*sin(phase)] (f16, row major and
// transposed) and V = [mag | phase] (f16, row major).
// ---------------------------------------------------------------------------
__global__ void prep_qkv(const float* __restrict__ mag, const float* __restrict__ phase,
                         _Float16* __restrict__ qk, _Float16* __restrict__ qkT,
                         _Float16* __restrict__ vv) {
  int idx = blockIdx.x * blockDim.x + threadIdx.x;   // N*D threads
  int row = idx >> 6;
  int d   = idx & 63;
  float m = mag[idx];
  float p = phase[idx];
  float sn, cs;
  __sincosf(p, &sn, &cs);
  _Float16 ch = (_Float16)(m * cs);
  _Float16 sh = (_Float16)(m * sn);
  qk[row * DQK + d]       = ch;
  qk[row * DQK + 64 + d]  = sh;
  qkT[(size_t)d        * N_NODES + row] = ch;
  qkT[(size_t)(64 + d) * N_NODES + row] = sh;
  vv[row * DV + d]        = (_Float16)m;
  vv[row * DV + 64 + d]   = (_Float16)p;
}

// ---------------------------------------------------------------------------
// Kernel 2: collapse Linear(edge_dim->hidden).sum(-1) to wcol[4] + bsum.
// ---------------------------------------------------------------------------
__global__ void edge_params(const float* __restrict__ W, const float* __restrict__ b,
                            float* __restrict__ params) {
  int t = threadIdx.x;
  if (t < 4) {
    float s = 0.f;
    for (int d = 0; d < D_HID; ++d) s += W[d * 4 + t];
    params[t] = s;
  } else if (t == 4) {
    float s = 0.f;
    for (int d = 0; d < D_HID; ++d) s += b[d];
    params[4] = s;
  }
}

// ---------------------------------------------------------------------------
// Kernel 3: scatter per-edge bias into dense [N,N] f16 bias matrix.
// f16 halves the dominant HBM term (memset + streaming read) vs f32.
// ---------------------------------------------------------------------------
__global__ void edge_scatter(const long long* __restrict__ ei,
                             const float* __restrict__ ea,
                             const float* __restrict__ params,
                             _Float16* __restrict__ bias) {
  int e = blockIdx.x * blockDim.x + threadIdx.x;
  if (e >= N_EDGES) return;
  long long src = ei[e];
  long long dst = ei[N_EDGES + e];
  float s = params[4];
  #pragma unroll
  for (int k = 0; k < 4; ++k) s += ea[e * 4 + k] * params[k];
  bias[(size_t)src * N_NODES + dst] = (_Float16)s;  // duplicates: last-writer, like .at[].set
}

// ---------------------------------------------------------------------------
// Kernel 4: flash attention. 8 waves/block, each wave owns 16 query rows and
// streams 32-column K/V tiles with online softmax. All B operands and bias
// gathers for a tile are issued as one batch before the WMMA chains so load
// latency is covered once per tile, not per WMMA.
// ---------------------------------------------------------------------------
__global__ void __launch_bounds__(256)
flash_attn(const _Float16* __restrict__ qk, const _Float16* __restrict__ qkT,
           const _Float16* __restrict__ vv, const _Float16* __restrict__ bias,
           float* __restrict__ out) {
  __shared__ _Float16 lds_p[8][16 * 32];   // per-wave P staging (C-layout -> A-layout)

  const int lane  = threadIdx.x & 31;
  const int wave  = threadIdx.x >> 5;
  const int ibase = blockIdx.x * 128 + wave * 16;
  const int mrow  = lane & 15;   // A-operand M row / C-layout N column
  const int hi16  = lane >> 4;   // half-wave selector

  // ---- Q in A-operand layout: 4 chunks of 16x32, two contiguous v8h segs each
  v16h Aq[4];
  {
    const _Float16* qrow = qk + (size_t)(ibase + mrow) * DQK + hi16 * 8;
    #pragma unroll
    for (int c = 0; c < 4; ++c) {
      v8h lo = *(const v8h*)(qrow + c * 32);
      v8h hh = *(const v8h*)(qrow + c * 32 + 16);
      Aq[c] = cat8(lo, hh);
    }
  }

  v8f O[8];
  #pragma unroll
  for (int t = 0; t < 8; ++t) O[t] = {};
  float m_i[8], l_i[8];                       // l_i: LANE-LOCAL partial denominator
  #pragma unroll
  for (int r = 0; r < 8; ++r) { m_i[r] = -__builtin_inff(); l_i[r] = 0.f; }

  for (int jb = 0; jb < N_NODES / 32; ++jb) {
    const int j0 = jb * 32;

    // ---- issue ALL K-tile B-operand loads up front (one clause, 16x b128)
    v16h Bk[8];
    #pragma unroll
    for (int kc = 0; kc < 4; ++kc) {
      const _Float16* bp = qk + (size_t)(j0 + mrow) * DQK + kc * 32 + hi16 * 8;
      Bk[kc * 2 + 0] = cat8(*(const v8h*)bp,        *(const v8h*)(bp + 16));
      Bk[kc * 2 + 1] = cat8(*(const v8h*)(bp + 16 * DQK), *(const v8h*)(bp + 16 * DQK + 16));
    }

    // ---- issue f16 bias gathers early; consumed only after the WMMA chains
    const _Float16* bptr = bias + (size_t)(ibase + hi16 * 8) * N_NODES + j0 + mrow;
    _Float16 bv0[8], bv1[8];
    #pragma unroll
    for (int r = 0; r < 8; ++r) {
      bv0[r] = bptr[(size_t)r * N_NODES];
      bv1[r] = bptr[(size_t)r * N_NODES + 16];
    }
    if (jb + 1 < N_NODES / 32) __builtin_prefetch(bptr + 32, 0, 1);  // global_prefetch_b8

    // ---- S[16,32] = Q @ K^T : two independent accumulation chains
    v8f S0 = {}, S1 = {};
    #pragma unroll
    for (int kc = 0; kc < 4; ++kc) {
      S0 = __builtin_amdgcn_wmma_f32_16x16x32_f16(false, Aq[kc], false, Bk[kc * 2 + 0], (short)0, S0, false, false);
      S1 = __builtin_amdgcn_wmma_f32_16x16x32_f16(false, Aq[kc], false, Bk[kc * 2 + 1], (short)0, S1, false, false);
    }
    #pragma unroll
    for (int r = 0; r < 8; ++r) {
      S0[r] += (float)bv0[r];
      S1[r] += (float)bv1[r];
    }

    // ---- online softmax: row-max butterfly only; sum stays lane-local
    float P0[8], P1[8], corr[8];
    #pragma unroll
    for (int r = 0; r < 8; ++r) {
      float vmax = fmaxf(S0[r], S1[r]);
      #pragma unroll
      for (int msk = 1; msk < 16; msk <<= 1)
        vmax = fmaxf(vmax, __shfl_xor(vmax, msk, 32));
      float mnew = fmaxf(m_i[r], vmax);
      corr[r] = __expf(m_i[r] - mnew);        // uniform across the half-wave
      P0[r]   = __expf(S0[r] - mnew);
      P1[r]   = __expf(S1[r] - mnew);
      l_i[r]  = l_i[r] * corr[r] + (P0[r] + P1[r]);   // lane-local
      m_i[r]  = mnew;
    }
    #pragma unroll
    for (int t = 0; t < 8; ++t)
      #pragma unroll
      for (int r = 0; r < 8; ++r)
        O[t][r] *= corr[r];

    // ---- P: C-layout f32 -> A-layout f16 via per-wave LDS round trip
    _Float16* lp = &lds_p[wave][0];
    #pragma unroll
    for (int r = 0; r < 8; ++r) {
      lp[(r + hi16 * 8) * 32 + mrow]      = (_Float16)P0[r];
      lp[(r + hi16 * 8) * 32 + 16 + mrow] = (_Float16)P1[r];
    }
    asm volatile("s_wait_dscnt 0" ::: "memory");   // CDNA5 split counter wait
    const _Float16* ap = lp + mrow * 32 + hi16 * 8;
    v16h Ap = cat8(*(const v8h*)ap, *(const v8h*)(ap + 16));

    // ---- O[16,128] += P[16,32] @ V[32,128] (V gathered: lane = column, elements = k in table order)
    const _Float16* vrow = vv + (size_t)j0 * DV + mrow;
    #pragma unroll
    for (int t = 0; t < 8; ++t) {
      v16h Bv;
      #pragma unroll
      for (int s = 0; s < 16; ++s)
        Bv[s] = vrow[(size_t)((s < 8 ? hi16 * 8 + s : 16 + hi16 * 8 + (s - 8))) * DV + t * 16];
      O[t] = __builtin_amdgcn_wmma_f32_16x16x32_f16(false, Ap, false, Bv, (short)0, O[t], false, false);
    }
  }

  // ---- deferred denominator reduction (once, not per tile)
  float inv[8];
  #pragma unroll
  for (int r = 0; r < 8; ++r) {
    float l = l_i[r];
    #pragma unroll
    for (int msk = 1; msk < 16; msk <<= 1)
      l += __shfl_xor(l, msk, 32);
    inv[r] = 1.0f / l;
  }

  // ---- normalize and store: cols 0..63 -> new_mag, 64..127 -> new_phase
  #pragma unroll
  for (int t = 0; t < 8; ++t) {
    #pragma unroll
    for (int r = 0; r < 8; ++r) {
      int   row  = ibase + hi16 * 8 + r;
      int   colv = t * 16 + mrow;
      float val  = O[t][r] * inv[r];
      if (t < 4) out[(size_t)row * 64 + colv] = val;
      else       out[(size_t)N_NODES * 64 + (size_t)row * 64 + (colv - 64)] = val;
    }
  }
}

// ---------------------------------------------------------------------------
// Workspace layout (bytes):
//   [0, 2M)          qk   row-major f16   N*128
//   [2M, 4M)         qkT  transposed f16  128*N
//   [4M, 6M)         v    row-major f16   N*128
//   [6M, 6M+256)     params (wcol[4], bsum)
//   [6M+256, +128MB) bias f16 N*N
// ---------------------------------------------------------------------------
extern "C" void kernel_launch(void* const* d_in, const int* in_sizes, int n_in,
                              void* d_out, int out_size, void* d_ws, size_t ws_size,
                              hipStream_t stream) {
  const float*     mag   = (const float*)d_in[0];
  const float*     phase = (const float*)d_in[1];
  const long long* ei    = (const long long*)d_in[2];   // int64 edge_index [2,E]
  const float*     ea    = (const float*)d_in[3];
  const float*     W     = (const float*)d_in[4];
  const float*     b     = (const float*)d_in[5];
  float*           out   = (float*)d_out;

  const size_t QK_BYTES = (size_t)N_NODES * DQK * sizeof(_Float16);  // 2 MiB
  char* ws = (char*)d_ws;
  _Float16* qk     = (_Float16*)(ws);
  _Float16* qkT    = (_Float16*)(ws + QK_BYTES);
  _Float16* vv     = (_Float16*)(ws + 2 * QK_BYTES);
  float*    params = (float*)(ws + 3 * QK_BYTES);
  _Float16* bias   = (_Float16*)(ws + 3 * QK_BYTES + 256);

  hipMemsetAsync(bias, 0, (size_t)N_NODES * N_NODES * sizeof(_Float16), stream);

  prep_qkv<<<(N_NODES * D_HID) / 256, 256, 0, stream>>>(mag, phase, qk, qkT, vv);
  edge_params<<<1, 64, 0, stream>>>(W, b, params);
  edge_scatter<<<N_EDGES / 256, 256, 0, stream>>>(ei, ea, params, bias);
  flash_attn<<<N_NODES / 128, 256, 0, stream>>>(qk, qkT, vv, bias, out);
}
